// MultiHeadBiasedAttentionADALN_MM_30588757082319
// MI455X (gfx1250) — matrix operand order repaired and checked by reading, not compiled, not run on hardware
//
#include <hip/hip_runtime.h>
#include <hip/hip_bf16.h>

typedef __attribute__((ext_vector_type(16))) __bf16 v16bf;
typedef __attribute__((ext_vector_type(8)))  __bf16 v8bf;
typedef __attribute__((ext_vector_type(8)))  float  v8f;

#define NTOK 768
#define DMODEL 768
#define NHEAD 12
#define DHEAD 64
#define DPAIR 128
#define DCOND 512

__device__ __forceinline__ v8f wmma_bf16(v16bf a, v16bf b, v8f c) {
  return __builtin_amdgcn_wmma_f32_16x16x32_bf16(false, a, false, b, (short)0, c,
                                                 false, false);
}

// Build an A fragment from two contiguous 8-element halves (k order this code uses for both A and B:
// elements 0..15 hold K = half*16..half*16+15, contiguous).
__device__ __forceinline__ v16bf a_frag(const __bf16* base /* row ptr + half*16 */) {
  v8bf lo = *(const v8bf*)(base);
  v8bf hi = *(const v8bf*)(base + 8);
  return __builtin_shufflevector(lo, hi, 0, 1, 2, 3, 4, 5, 6, 7,
                                 8, 9, 10, 11, 12, 13, 14, 15);
}

// MT x 1 strip of 16x16 tiles: A(bf16,[M,K]) x W^T(bf16,[N,K]). One wave.
// B fragment for lane(col nc, half): 16 consecutive K values -> one v16bf load.
// unroll 1: each iteration already holds 4 WMMAs + 10 b128 loads of ILP;
// deeper pipelining spills VGPRs.
template <int K, int MT>
__device__ __forceinline__ void gemm_tiles(const __bf16* __restrict__ A, int lda,
                                           const __bf16* __restrict__ Wt,
                                           int m0, int n0, v8f* acc) {
  const int lane = threadIdx.x & 31;
  const int mrow = lane & 15;
  const int half = lane >> 4;
  const __bf16* wrow = Wt + (size_t)(n0 + (lane & 15)) * K + half * 16;
#pragma unroll 1
  for (int k0 = 0; k0 < K; k0 += 32) {
    v16bf b = *(const v16bf*)(wrow + k0);
    if (k0 + 32 < K) __builtin_prefetch(wrow + k0 + 32, 0, 1);
#pragma unroll
    for (int t = 0; t < MT; ++t) {
      const __bf16* ap = A + (size_t)(m0 + t * 16 + mrow) * lda + k0 + half * 16;
      acc[t] = wmma_bf16(a_frag(ap), b, acc[t]);
    }
  }
}

// Dual-weight variant: same A feeds two weight streams (shares A fragments).
template <int K, int MT>
__device__ __forceinline__ void gemm_tiles_dual(const __bf16* __restrict__ A, int lda,
                                                const __bf16* __restrict__ Wt0,
                                                const __bf16* __restrict__ Wt1,
                                                int m0, int n0, v8f* acc0, v8f* acc1) {
  const int lane = threadIdx.x & 31;
  const int mrow = lane & 15;
  const int half = lane >> 4;
  const __bf16* wrow0 = Wt0 + (size_t)(n0 + (lane & 15)) * K + half * 16;
  const __bf16* wrow1 = Wt1 + (size_t)(n0 + (lane & 15)) * K + half * 16;
#pragma unroll 1
  for (int k0 = 0; k0 < K; k0 += 32) {
    v16bf b0 = *(const v16bf*)(wrow0 + k0);
    v16bf b1 = *(const v16bf*)(wrow1 + k0);
#pragma unroll
    for (int t = 0; t < MT; ++t) {
      const __bf16* ap = A + (size_t)(m0 + t * 16 + mrow) * lda + k0 + half * 16;
      v16bf a = a_frag(ap);
      acc0[t] = wmma_bf16(a, b0, acc0[t]);
      acc1[t] = wmma_bf16(a, b1, acc1[t]);
    }
  }
}

__device__ __forceinline__ float block_reduce(float v, float* sbuf) {
  __syncthreads();
#pragma unroll
  for (int off = 16; off; off >>= 1) v += __shfl_xor(v, off, 32);
  const int tid = threadIdx.x;
  if ((tid & 31) == 0) sbuf[tid >> 5] = v;
  __syncthreads();
  float r = 0.f;
  if (tid < 8) r = sbuf[tid];
#pragma unroll
  for (int off = 4; off; off >>= 1) r += __shfl_xor(r, off, 32);
  if (tid == 0) sbuf[0] = r;
  __syncthreads();
  return sbuf[0];
}

// ---------- Kernel 0: fp32 [K,N] -> bf16 [N,K] transpose-convert ----------
__global__ __launch_bounds__(256) void k_cvt_t(const float* __restrict__ src,
                                               __bf16* __restrict__ dst, int K, int N) {
  __shared__ __bf16 tile[32][33];
  const int bx = blockIdx.x * 32;  // N offset
  const int by = blockIdx.y * 32;  // K offset
  const int tx = threadIdx.x & 31, ty = threadIdx.x >> 5;  // 256 thr: ty 0..7
#pragma unroll
  for (int r = ty; r < 32; r += 8)
    tile[r][tx] = (__bf16)src[(size_t)(by + r) * N + bx + tx];
  __syncthreads();
#pragma unroll
  for (int r = ty; r < 32; r += 8)
    dst[(size_t)(bx + r) * K + by + tx] = tile[tx][r];
}

// ---------- Kernel 1: LN(x), LN(cond)*g+b (bf16), raw cond (bf16) ----------
__global__ __launch_bounds__(256) void k_prepare(
    const float* __restrict__ x, const float* __restrict__ cond,
    const float* __restrict__ cg, const float* __restrict__ cb,
    float* __restrict__ lnx, __bf16* __restrict__ cn_bf, __bf16* __restrict__ cond_bf) {
  __shared__ float sbuf[8];
  const int i = blockIdx.x;
  const int tid = threadIdx.x;
  float s = 0.f, s2 = 0.f;
  for (int j = tid; j < DMODEL; j += blockDim.x) {
    float v = x[i * DMODEL + j]; s += v; s2 += v * v;
  }
  s = block_reduce(s, sbuf);
  s2 = block_reduce(s2, sbuf);
  float m = s * (1.f / DMODEL);
  float r = rsqrtf(s2 * (1.f / DMODEL) - m * m + 1e-5f);
  for (int j = tid; j < DMODEL; j += blockDim.x)
    lnx[i * DMODEL + j] = (x[i * DMODEL + j] - m) * r;

  s = 0.f; s2 = 0.f;
  for (int j = tid; j < DCOND; j += blockDim.x) {
    float v = cond[i * DCOND + j]; s += v; s2 += v * v;
  }
  s = block_reduce(s, sbuf);
  s2 = block_reduce(s2, sbuf);
  m = s * (1.f / DCOND);
  r = rsqrtf(s2 * (1.f / DCOND) - m * m + 1e-5f);
  for (int j = tid; j < DCOND; j += blockDim.x) {
    float v = cond[i * DCOND + j];
    cn_bf[i * DCOND + j] = (__bf16)((v - m) * r * cg[j] + cb[j]);
    cond_bf[i * DCOND + j] = (__bf16)v;
  }
}

// ---------- Kernel 2: h = LN(x)*sigmoid(cn@Wg+bg) + cn@Wb, masked ----------
__global__ __launch_bounds__(32) void k_adaln(
    const __bf16* __restrict__ cn_bf, const __bf16* __restrict__ WgT,
    const float* __restrict__ bg, const __bf16* __restrict__ WbT,
    const float* __restrict__ lnx, const unsigned char* __restrict__ mask,
    __bf16* __restrict__ h_bf) {
  const int m0 = blockIdx.y * 64, n0 = blockIdx.x * 16;
  v8f ag[4] = {}, ab[4] = {};
  gemm_tiles_dual<DCOND, 4>(cn_bf, DCOND, WgT, WbT, m0, n0, ag, ab);
  const int lane = threadIdx.x & 31;
  const int nn = n0 + (lane & 15);
  const int half = lane >> 4;
#pragma unroll
  for (int t = 0; t < 4; ++t)
#pragma unroll
    for (int r = 0; r < 8; ++r) {
      const int i = m0 + t * 16 + r + 8 * half;
      float gate = 1.f / (1.f + __expf(-(ag[t][r] + bg[nn])));
      float hv = lnx[i * DMODEL + nn] * gate + ab[t][r];
      if (!mask[i]) hv = 0.f;
      h_bf[i * DMODEL + nn] = (__bf16)hv;
    }
}

// ---------- Kernel 3: q/k/v = h @ W + b ----------
__global__ __launch_bounds__(32) void k_qkv(
    const __bf16* __restrict__ h_bf,
    const __bf16* __restrict__ WqT, const float* __restrict__ bq,
    const __bf16* __restrict__ WkT, const float* __restrict__ bk,
    const __bf16* __restrict__ WvT, const float* __restrict__ bv,
    float* __restrict__ qf, float* __restrict__ kf, float* __restrict__ vf) {
  const int m0 = blockIdx.y * 64, n0 = blockIdx.x * 16;
  const __bf16* W; const float* bb; float* out;
  if (blockIdx.z == 0)      { W = WqT; bb = bq; out = qf; }
  else if (blockIdx.z == 1) { W = WkT; bb = bk; out = kf; }
  else                      { W = WvT; bb = bv; out = vf; }
  v8f c[4] = {};
  gemm_tiles<DMODEL, 4>(h_bf, DMODEL, W, m0, n0, c);
  const int lane = threadIdx.x & 31;
  const int nn = n0 + (lane & 15);
  const int half = lane >> 4;
#pragma unroll
  for (int t = 0; t < 4; ++t)
#pragma unroll
    for (int r = 0; r < 8; ++r)
      out[(m0 + t * 16 + r + 8 * half) * DMODEL + nn] = c[t][r] + bb[nn];
}

// ---------- Kernel 4: per-head LN of q,k; pack q/k head-major, V transposed ----------
__global__ __launch_bounds__(128) void k_qkln(
    const float* __restrict__ qf, const float* __restrict__ kf,
    const float* __restrict__ vf,
    const float* __restrict__ qg, const float* __restrict__ qb,
    const float* __restrict__ kg, const float* __restrict__ kb,
    __bf16* __restrict__ q_bf, __bf16* __restrict__ k_bf, __bf16* __restrict__ v_t) {
  const int lane = threadIdx.x & 31, wid = threadIdx.x >> 5;
  const int g = blockIdx.x * 4 + wid;
  const int i = g / NHEAD, h = g % NHEAD;
  const int d0 = lane * 2;

  const float* qrow = qf + i * DMODEL + h * DHEAD;
  float a = qrow[d0], b = qrow[d0 + 1];
  float s = a + b, s2 = a * a + b * b;
#pragma unroll
  for (int off = 16; off; off >>= 1) { s += __shfl_xor(s, off, 32); s2 += __shfl_xor(s2, off, 32); }
  float m = s * (1.f / DHEAD);
  float r = rsqrtf(s2 * (1.f / DHEAD) - m * m + 1e-5f);
  __bf16* qo = q_bf + (h * NTOK + i) * DHEAD;
  qo[d0]     = (__bf16)((a - m) * r * qg[d0] + qb[d0]);
  qo[d0 + 1] = (__bf16)((b - m) * r * qg[d0 + 1] + qb[d0 + 1]);

  const float* krow = kf + i * DMODEL + h * DHEAD;
  a = krow[d0]; b = krow[d0 + 1];
  s = a + b; s2 = a * a + b * b;
#pragma unroll
  for (int off = 16; off; off >>= 1) { s += __shfl_xor(s, off, 32); s2 += __shfl_xor(s2, off, 32); }
  m = s * (1.f / DHEAD);
  r = rsqrtf(s2 * (1.f / DHEAD) - m * m + 1e-5f);
  __bf16* ko = k_bf + (h * NTOK + i) * DHEAD;
  ko[d0]     = (__bf16)((a - m) * r * kg[d0] + kb[d0]);
  ko[d0 + 1] = (__bf16)((b - m) * r * kg[d0 + 1] + kb[d0 + 1]);

  // V stored transposed per head: v_t[h][d][tok]
  const float* vrow = vf + i * DMODEL + h * DHEAD;
  __bf16* vo = v_t + ((size_t)h * DHEAD + d0) * NTOK + i;
  vo[0]    = (__bf16)vrow[d0];
  vo[NTOK] = (__bf16)vrow[d0 + 1];
}

// ---------- Kernel 5: bias[h][i][j] = (LN(pair[i][j])*g+b) @ Wpb ----------
__global__ __launch_bounds__(128) void k_pairbias(
    const float* __restrict__ pair, const float* __restrict__ pg,
    const float* __restrict__ pb, const float* __restrict__ Wpb,
    float* __restrict__ biasb) {
  const int lane = threadIdx.x & 31, wid = threadIdx.x >> 5;
  const long pos = (long)blockIdx.x * 4 + wid;
  const int i = (int)(pos / NTOK), j = (int)(pos % NTOK);
  const float4 v = *(const float4*)(pair + pos * DPAIR + lane * 4);
  float s = v.x + v.y + v.z + v.w;
  float s2 = v.x * v.x + v.y * v.y + v.z * v.z + v.w * v.w;
#pragma unroll
  for (int off = 16; off; off >>= 1) { s += __shfl_xor(s, off, 32); s2 += __shfl_xor(s2, off, 32); }
  const float m = s * (1.f / DPAIR);
  const float r = rsqrtf(s2 * (1.f / DPAIR) - m * m + 1e-5f);
  float vv[4] = { v.x, v.y, v.z, v.w };
  float acc[NHEAD];
#pragma unroll
  for (int h = 0; h < NHEAD; ++h) acc[h] = 0.f;
#pragma unroll
  for (int c = 0; c < 4; ++c) {
    const int cgl = lane * 4 + c;
    const float t = (vv[c] - m) * r * pg[cgl] + pb[cgl];
#pragma unroll
    for (int h = 0; h < NHEAD; ++h) acc[h] += t * Wpb[cgl * NHEAD + h];
  }
#pragma unroll
  for (int h = 0; h < NHEAD; ++h) {
    float t = acc[h];
#pragma unroll
    for (int off = 16; off; off >>= 1) t += __shfl_xor(t, off, 32);
    acc[h] = t;
  }
  if (lane < NHEAD)
    biasb[((long)lane * NTOK + i) * NTOK + j] = acc[lane];
}

// ---------- Kernel 6: attention (scores -> softmax -> AV), all WMMA ----------
#define ATT_LDS_PER_WAVE (49152 + 4096)
__global__ __launch_bounds__(128) void k_attn(
    const __bf16* __restrict__ q_bf, const __bf16* __restrict__ k_bf,
    const __bf16* __restrict__ v_t, const float* __restrict__ biasb,
    const unsigned char* __restrict__ mask, __bf16* __restrict__ o_bf) {
  extern __shared__ char smem[];
  const int lane = threadIdx.x & 31, wid = threadIdx.x >> 5;
  const int g = blockIdx.x * 4 + wid;
  const int head = g / (NTOK / 16);
  const int i0 = (g % (NTOK / 16)) * 16;
  float* scores = (float*)(smem + wid * ATT_LDS_PER_WAVE);          // [16][768] f32
  __bf16* stag = (__bf16*)(smem + wid * ATT_LDS_PER_WAVE + 49152);  // 4KB staging
  __bf16* qtile = stag;          // 16x64 row-major [tok][d]
  __bf16* ktile = stag + 1024;   // 16x64 row-major [tok][d]
  __bf16* vtile = stag;          // 64x32 [d][tok] (AV phase)
  const int n = lane & 15, half = lane >> 4;

  // stage q tile (b128 global -> b128 LDS), build 2 A fragments
  for (int c = lane; c < 128; c += 32) {
    const int row = c >> 3, col = (c & 7) * 8;
    *(uint4*)&qtile[c * 8] =
        *(const uint4*)&q_bf[(head * NTOK + i0 + row) * DHEAD + col];
  }
  __syncthreads();
  v16bf aq0 = a_frag(qtile + n * DHEAD + half * 16);
  v16bf aq1 = a_frag(qtile + n * DHEAD + 32 + half * 16);
  __syncthreads();

  const float rs = 0.125f;  // 1/sqrt(64)
#pragma unroll 1
  for (int jt = 0; jt < NTOK / 16; ++jt) {
    const int j0 = jt * 16;
    for (int c = lane; c < 128; c += 32) {
      const int row = c >> 3, col = (c & 7) * 8;
      *(uint4*)&ktile[c * 8] =
          *(const uint4*)&k_bf[(head * NTOK + j0 + row) * DHEAD + col];
    }
    if (jt + 1 < NTOK / 16)
      __builtin_prefetch(&k_bf[(head * NTOK + j0 + 16) * DHEAD], 0, 1);
    __syncthreads();
    // K^T B fragments: contiguous 32B runs in LDS
    v16bf b0 = *(const v16bf*)(ktile + n * DHEAD + half * 16);
    v16bf b1 = *(const v16bf*)(ktile + n * DHEAD + 32 + half * 16);
    v8f c = {};
    c = wmma_bf16(aq0, b0, c);
    c = wmma_bf16(aq1, b1, c);
    const int jc = j0 + n;
    const float mj = mask[jc] ? 1.f : 0.f;
#pragma unroll
    for (int r = 0; r < 8; ++r) {
      const int il = r + 8 * half;
      const float mi = mask[i0 + il] ? 1.f : 0.f;
      float val = c[r] * rs + biasb[((long)head * NTOK + (i0 + il)) * NTOK + jc];
      if (mi * mj <= 0.f) val = -1e9f;
      scores[il * NTOK + jc] = val;
    }
    __syncthreads();
  }

  // softmax: one lane per row, in-place fp32 exp then bf16 normalized probs
  if (lane < 16) {
    volatile float* srow = scores + lane * NTOK;
    float mx = -1e30f;
    for (int j = 0; j < NTOK; ++j) mx = fmaxf(mx, srow[j]);
    float sum = 0.f;
    for (int j = 0; j < NTOK; ++j) { float e = __expf(srow[j] - mx); srow[j] = e; sum += e; }
    const float inv = 1.f / sum;
    volatile __bf16* prow = (volatile __bf16*)(scores + lane * NTOK);
    for (int j = 0; j < NTOK; ++j) prow[j] = (__bf16)(srow[j] * inv);
  }
  __syncthreads();

  // AV: o = P (16x768 bf16) @ V (768x64); V staged as [d][tok] so B frags
  // are contiguous v16bf LDS loads.
  const __bf16* probs = (const __bf16*)scores;  // row m at probs + m*1536
  v8f o0 = {}, o1 = {}, o2 = {}, o3 = {};
#pragma unroll 1
  for (int kt = 0; kt < NTOK / 32; ++kt) {
    const int j0 = kt * 32;
    __syncthreads();
    for (int c = lane; c < 256; c += 32) {
      const int row = c >> 2, col = (c & 3) * 8;  // 64 d-rows x 32 toks
      *(uint4*)&vtile[row * 32 + col] =
          *(const uint4*)&v_t[((size_t)head * DHEAD + row) * NTOK + j0 + col];
    }
    if (kt + 1 < NTOK / 32)
      __builtin_prefetch(&v_t[(size_t)head * DHEAD * NTOK + j0 + 32], 0, 1);
    __syncthreads();
    v16bf a = a_frag(probs + n * (2 * NTOK) + j0 + half * 16);
    v16bf bv0 = *(const v16bf*)(vtile + (0 * 16 + n) * 32 + half * 16);
    v16bf bv1 = *(const v16bf*)(vtile + (1 * 16 + n) * 32 + half * 16);
    v16bf bv2 = *(const v16bf*)(vtile + (2 * 16 + n) * 32 + half * 16);
    v16bf bv3 = *(const v16bf*)(vtile + (3 * 16 + n) * 32 + half * 16);
    o0 = wmma_bf16(a, bv0, o0);
    o1 = wmma_bf16(a, bv1, o1);
    o2 = wmma_bf16(a, bv2, o2);
    o3 = wmma_bf16(a, bv3, o3);
  }
#pragma unroll
  for (int r = 0; r < 8; ++r) {
    const int i = i0 + r + 8 * half;
    o_bf[i * DMODEL + head * DHEAD + 0 * 16 + n] = (__bf16)o0[r];
    o_bf[i * DMODEL + head * DHEAD + 1 * 16 + n] = (__bf16)o1[r];
    o_bf[i * DMODEL + head * DHEAD + 2 * 16 + n] = (__bf16)o2[r];
    o_bf[i * DMODEL + head * DHEAD + 3 * 16 + n] = (__bf16)o3[r];
  }
}

// ---------- Kernel 7: out = (o@Wo + bo) * sigmoid(cond@sW + sb) * mask ----------
__global__ __launch_bounds__(32) void k_output(
    const __bf16* __restrict__ o_bf, const __bf16* __restrict__ WoT,
    const float* __restrict__ bo, const __bf16* __restrict__ cond_bf,
    const __bf16* __restrict__ sWT, const float* __restrict__ sb,
    const unsigned char* __restrict__ mask, float* __restrict__ out) {
  const int m0 = blockIdx.y * 64, n0 = blockIdx.x * 16;
  v8f c1[4] = {}, c2[4] = {};
  gemm_tiles<DMODEL, 4>(o_bf, DMODEL, WoT, m0, n0, c1);
  gemm_tiles<DCOND, 4>(cond_bf, DCOND, sWT, m0, n0, c2);
  const int lane = threadIdx.x & 31;
  const int nn = n0 + (lane & 15);
  const int half = lane >> 4;
#pragma unroll
  for (int t = 0; t < 4; ++t)
#pragma unroll
    for (int r = 0; r < 8; ++r) {
      const int i = m0 + t * 16 + r + 8 * half;
      const float mf = mask[i] ? 1.f : 0.f;
      const float gate = 1.f / (1.f + __expf(-(c2[t][r] + sb[nn])));
      out[i * DMODEL + nn] = (c1[t][r] + bo[nn]) * gate * mf;
    }
}

extern "C" void kernel_launch(void* const* d_in, const int* in_sizes, int n_in,
                              void* d_out, int out_size, void* d_ws, size_t ws_size,
                              hipStream_t stream) {
  (void)in_sizes; (void)n_in; (void)out_size; (void)ws_size;
  const float* x      = (const float*)d_in[0];
  const float* pair   = (const float*)d_in[1];
  const float* cond   = (const float*)d_in[2];
  const float* acg    = (const float*)d_in[3];
  const float* acb    = (const float*)d_in[4];
  const float* Wg     = (const float*)d_in[5];
  const float* bg     = (const float*)d_in[6];
  const float* Wb     = (const float*)d_in[7];
  const float* Wq     = (const float*)d_in[8];
  const float* bq     = (const float*)d_in[9];
  const float* Wk     = (const float*)d_in[10];
  const float* bk     = (const float*)d_in[11];
  const float* Wv     = (const float*)d_in[12];
  const float* bv     = (const float*)d_in[13];
  const float* qg     = (const float*)d_in[14];
  const float* qb     = (const float*)d_in[15];
  const float* kg     = (const float*)d_in[16];
  const float* kb     = (const float*)d_in[17];
  const float* pg     = (const float*)d_in[18];
  const float* pb     = (const float*)d_in[19];
  const float* Wpb    = (const float*)d_in[20];
  const float* Wo     = (const float*)d_in[21];
  const float* bo     = (const float*)d_in[22];
  const float* sW     = (const float*)d_in[23];
  const float* sb     = (const float*)d_in[24];
  const unsigned char* mask = (const unsigned char*)d_in[25];
  float* out = (float*)d_out;

  // workspace carve-out (256B aligned)
  char* ws = (char*)d_ws;
  size_t off = 0;
  auto carve = [&](size_t bytes) { char* p = ws + off; off = (off + bytes + 255) & ~(size_t)255; return p; };
  __bf16* cn_bf   = (__bf16*)carve((size_t)NTOK * DCOND * 2);
  __bf16* cond_bf = (__bf16*)carve((size_t)NTOK * DCOND * 2);
  float*  lnx     = (float*) carve((size_t)NTOK * DMODEL * 4);
  __bf16* h_bf    = (__bf16*)carve((size_t)NTOK * DMODEL * 2);
  float*  qf      = (float*) carve((size_t)NTOK * DMODEL * 4);
  float*  kf      = (float*) carve((size_t)NTOK * DMODEL * 4);
  float*  vf      = (float*) carve((size_t)NTOK * DMODEL * 4);
  __bf16* q_bf    = (__bf16*)carve((size_t)NTOK * DMODEL * 2);
  __bf16* k_bf    = (__bf16*)carve((size_t)NTOK * DMODEL * 2);
  __bf16* v_t     = (__bf16*)carve((size_t)NTOK * DMODEL * 2);
  float*  biasb   = (float*) carve((size_t)NHEAD * NTOK * NTOK * 4);
  __bf16* o_bf    = (__bf16*)carve((size_t)NTOK * DMODEL * 2);
  // transposed bf16 weights [N,K]
  __bf16* WgT = (__bf16*)carve((size_t)DCOND * DMODEL * 2);
  __bf16* WbT = (__bf16*)carve((size_t)DCOND * DMODEL * 2);
  __bf16* WqT = (__bf16*)carve((size_t)DMODEL * DMODEL * 2);
  __bf16* WkT = (__bf16*)carve((size_t)DMODEL * DMODEL * 2);
  __bf16* WvT = (__bf16*)carve((size_t)DMODEL * DMODEL * 2);
  __bf16* WoT = (__bf16*)carve((size_t)DMODEL * DMODEL * 2);
  __bf16* sWT = (__bf16*)carve((size_t)DCOND * DMODEL * 2);

  const dim3 gCD(DMODEL / 32, DCOND / 32);   // src [DCOND, DMODEL]
  const dim3 gDD(DMODEL / 32, DMODEL / 32);  // src [DMODEL, DMODEL]
  k_cvt_t<<<gCD, 256, 0, stream>>>(Wg, WgT, DCOND, DMODEL);
  k_cvt_t<<<gCD, 256, 0, stream>>>(Wb, WbT, DCOND, DMODEL);
  k_cvt_t<<<gDD, 256, 0, stream>>>(Wq, WqT, DMODEL, DMODEL);
  k_cvt_t<<<gDD, 256, 0, stream>>>(Wk, WkT, DMODEL, DMODEL);
  k_cvt_t<<<gDD, 256, 0, stream>>>(Wv, WvT, DMODEL, DMODEL);
  k_cvt_t<<<gDD, 256, 0, stream>>>(Wo, WoT, DMODEL, DMODEL);
  k_cvt_t<<<gCD, 256, 0, stream>>>(sW, sWT, DCOND, DMODEL);

  k_prepare<<<NTOK, 256, 0, stream>>>(x, cond, acg, acb, lnx, cn_bf, cond_bf);
  k_adaln<<<dim3(DMODEL / 16, NTOK / 64), 32, 0, stream>>>(cn_bf, WgT, bg, WbT, lnx, mask, h_bf);
  k_qkv<<<dim3(DMODEL / 16, NTOK / 64, 3), 32, 0, stream>>>(h_bf, WqT, bq, WkT, bk, WvT, bv,
                                                            qf, kf, vf);
  k_qkln<<<(NTOK * NHEAD) / 4, 128, 0, stream>>>(qf, kf, vf, qg, qb, kg, kb, q_bf, k_bf, v_t);
  k_pairbias<<<(NTOK * NTOK) / 4, 128, 0, stream>>>(pair, pg, pb, Wpb, biasb);

  const int attn_lds = 4 * ATT_LDS_PER_WAVE;  // 212992 B (< 320 KB WGP LDS)
  hipFuncSetAttribute((const void*)k_attn, hipFuncAttributeMaxDynamicSharedMemorySize, attn_lds);
  k_attn<<<(NHEAD * (NTOK / 16)) / 4, 128, attn_lds, stream>>>(q_bf, k_bf, v_t, biasb, mask, o_bf);

  k_output<<<dim3(DMODEL / 16, NTOK / 64), 32, 0, stream>>>(o_bf, WoT, bo, cond_bf, sWT, sb,
                                                            mask, out);
}
